// AutoCorrelation_72078141162123
// MI455X (gfx1250) — matrix operand order repaired and checked by reading, not compiled, not run on hardware
//
#include <hip/hip_runtime.h>
#include <hip/hip_bf16.h>
#include <stdint.h>

// Problem constants (match reference setup_inputs)
#define B_    8
#define L_    4096
#define D_    512
#define TOPK_ 8
#define BL_   (B_ * L_)

typedef __attribute__((ext_vector_type(16))) __bf16 v16bf;
typedef __attribute__((ext_vector_type(8)))  float  v8f;
typedef __attribute__((ext_vector_type(4))) unsigned int u32x4;
typedef __attribute__((ext_vector_type(8))) int i32x8;
typedef __attribute__((ext_vector_type(4))) int i32x4;

// ---------------- bf16 <-> f32 helpers (round-to-nearest-even) ----------------
__device__ __forceinline__ unsigned short f2bf(float f) {
  union { float f; unsigned u; } x; x.f = f;
  unsigned r = x.u + 0x7FFFu + ((x.u >> 16) & 1u);
  return (unsigned short)(r >> 16);
}
__device__ __forceinline__ float bf2f(unsigned short h) {
  union { unsigned u; float f; } x; x.u = ((unsigned)h) << 16;
  return x.f;
}

// ---------------- WMMA fragment loaders --------------------------------------
// A 16x32 bf16 (wave32): lane m (0..15): row m, K=0..15; lane m+16:
// row m, K=16..31.
// B 32x16 bf16: lane n (0..15): col n, K=0..15; lane n+16: col n, K=16..31.
union FragU { v16bf v; uint4 u[2]; };

__device__ __forceinline__ v16bf load_frag_a(const unsigned short* __restrict__ A,
                                             int ld, int row0, int kk) {
  int lane = threadIdx.x & 31;
  int m = lane & 15, half = lane >> 4;
  const unsigned short* p = A + (size_t)(row0 + m) * ld + kk + half * 16;
  FragU r;
  r.u[0] = *(const uint4*)(p);
  r.u[1] = *(const uint4*)(p + 8);
  return r.v;
}

__device__ __forceinline__ v16bf load_frag_b(const unsigned short* __restrict__ Wm,
                                             int ld, int col0, int kk) {
  int lane = threadIdx.x & 31;
  int n = lane & 15, half = lane >> 4;
  const unsigned short* p = Wm + (size_t)(col0 + n) * ld + kk + half * 16;
  FragU r;
  r.u[0] = *(const uint4*)(p);
  r.u[1] = *(const uint4*)(p + 8);
  return r.v;
}

__device__ __forceinline__ v8f wmma_bf16(v16bf a, v16bf b, v8f c) {
  return __builtin_amdgcn_wmma_f32_16x16x32_bf16(false, a, false, b,
                                                 (short)0, c, false, false);
}

// ---------------- elementwise utility kernels --------------------------------
__global__ void cvt_f32_bf16_kernel(const float* __restrict__ src,
                                    unsigned short* __restrict__ dst, int n) {
  for (int i = blockIdx.x * blockDim.x + threadIdx.x; i < n;
       i += gridDim.x * blockDim.x)
    dst[i] = f2bf(src[i]);
}

__global__ void zero_f32_kernel(float* __restrict__ p, int n) {
  for (int i = blockIdx.x * blockDim.x + threadIdx.x; i < n;
       i += gridDim.x * blockDim.x)
    p[i] = 0.0f;
}

// ---------------- GEMM:  C[M,N] = A[M,Kd] * W[N,Kd]^T + bias -----------------
// 256 threads = 8 waves; workgroup tile 128(M) x 64(N); wave tile 32x32.
template <int OUT_F32>
__global__ __launch_bounds__(256) void gemm_bf16_kernel(
    const unsigned short* __restrict__ A, const unsigned short* __restrict__ W,
    const float* __restrict__ bias, void* __restrict__ Cout,
    int M, int N, int Kd) {
  int wid = threadIdx.x >> 5;
  int wm = wid & 3, wn = wid >> 2;              // 4 x 2 wave grid
  int row0 = blockIdx.x * 128 + wm * 32;
  int col0 = blockIdx.y * 64 + wn * 32;

  v8f acc[2][2] = {};
  for (int kk = 0; kk < Kd; kk += 32) {
    v16bf a0 = load_frag_a(A, Kd, row0,      kk);
    v16bf a1 = load_frag_a(A, Kd, row0 + 16, kk);
    v16bf b0 = load_frag_b(W, Kd, col0,      kk);
    v16bf b1 = load_frag_b(W, Kd, col0 + 16, kk);
    acc[0][0] = wmma_bf16(a0, b0, acc[0][0]);
    acc[0][1] = wmma_bf16(a0, b1, acc[0][1]);
    acc[1][0] = wmma_bf16(a1, b0, acc[1][0]);
    acc[1][1] = wmma_bf16(a1, b1, acc[1][1]);
  }

  int lane = threadIdx.x & 31;
  int n = lane & 15, rbase = (lane >> 4) * 8;
#pragma unroll
  for (int i = 0; i < 2; ++i) {
#pragma unroll
    for (int j = 0; j < 2; ++j) {
      int col = col0 + j * 16 + n;
      float bv = bias[col];
#pragma unroll
      for (int r = 0; r < 8; ++r) {
        int row = row0 + i * 16 + rbase + r;
        float val = acc[i][j][r] + bv;
        if (OUT_F32)
          ((float*)Cout)[(size_t)row * N + col] = val;
        else
          ((unsigned short*)Cout)[(size_t)row * N + col] = f2bf(val);
      }
    }
  }
}

// ---------------- Gram-matrix autocorrelation with TDM staging ---------------
// mean_corr[b,tau] = (1/D) * sum_t <Q[b,t,:], K[b,(t-tau)%L,:]>
// K is staged into LDS in 128(rows) x 128(k) bf16 chunks by the Tensor Data
// Mover, double-buffered, with 16B padding every 256B row so the ds_read_b128
// B-fragment gathers are bank-conflict-free.
#define SCH_  128                              // K rows per staged chunk
#define KCH_  128                              // k elements per staged chunk
#define ROWB_ 272                              // 256B data + 16B pad per row
#define NCH_  ((L_ / SCH_) * (D_ / KCH_))      // 128 chunk loads per WG

__device__ __forceinline__ void tdm_load_k_chunk(const unsigned short* Kb,
                                                 unsigned lds_off,
                                                 int scI, int kcI) {
  unsigned long long gaddr =
      (unsigned long long)(size_t)Kb +
      ((unsigned long long)scI * SCH_ * D_ + (unsigned long long)kcI * KCH_) * 2ull;
  u32x4 g0;
  g0[0] = 1u;                                        // count=1 (valid D#)
  g0[1] = lds_off;                                   // LDS byte address
  g0[2] = (unsigned)(gaddr & 0xFFFFFFFFull);         // global_addr lo
  g0[3] = (unsigned)((gaddr >> 32) & 0x01FFFFFFull)  // global_addr hi
        | 0x80000000u;                               // type=2 (image)
  i32x8 g1;
  g1[0] = (1 << 16)                                  // data_size = 2 bytes
        | (1 << 20)                                  // pad_enable
        | (5 << 22)                                  // pad_interval = 64 DW
        | (3 << 25);                                 // pad_amount  = 4 DW (16B)
  g1[1] = (int)((unsigned)(D_ & 0xFFFF) << 16);      // tensor_dim0 = 512
  g1[2] = (int)((D_ >> 16) | ((L_ & 0xFFFF) << 16)); // dim0 hi | dim1 lo (4096)
  g1[3] = (int)((L_ >> 16) | (KCH_ << 16));          // dim1 hi | tile_dim0 = 128
  g1[4] = (int)SCH_;                                 // tile_dim1 = 128, dim2 = 0
  g1[5] = (int)D_;                                   // tensor_dim0_stride = 512
  g1[6] = 0;
  g1[7] = 0;
  i32x4 gz = {0, 0, 0, 0};
#if defined(__clang_major__) && (__clang_major__ >= 23)
  i32x8 gz8 = {0, 0, 0, 0, 0, 0, 0, 0};
  __builtin_amdgcn_tensor_load_to_lds(g0, g1, gz, gz, gz8, 0);
#else
  __builtin_amdgcn_tensor_load_to_lds(g0, g1, gz, gz, 0);
#endif
}

__device__ __forceinline__ v16bf load_frag_b_lds(const unsigned char* buf,
                                                 int stl, int ks) {
  int lane = threadIdx.x & 31;
  int nn = lane & 15, half = lane >> 4;
  const unsigned char* p = buf + (stl * 16 + nn) * ROWB_ + ks * 64 + half * 32;
  FragU r;
  r.u[0] = *(const uint4*)(p);
  r.u[1] = *(const uint4*)(p + 16);
  return r.v;
}

__global__ __launch_bounds__(256) void corr_kernel(
    const unsigned short* __restrict__ Q, const unsigned short* __restrict__ K,
    float* __restrict__ mean_corr) {
  __shared__ float sc[L_];                                        // 16 KB
  __shared__ __align__(16) unsigned char kstage[2][SCH_ * ROWB_]; // 2 x 34 KB
  for (int i = threadIdx.x; i < L_; i += 256) sc[i] = 0.0f;

  int b = blockIdx.y;
  const unsigned short* Qb = Q + (size_t)b * L_ * D_;
  const unsigned short* Kb = K + (size_t)b * L_ * D_;

  int wid = threadIdx.x >> 5;
  int mt  = wid & 3;                 // which 16-row M tile
  int par = wid >> 2;                // s-tile parity within chunk
  int t_tile = blockIdx.x * 64 + mt * 16;
  int lane = threadIdx.x & 31;
  int n = lane & 15, rbase = (lane >> 4) * 8;

  // All 16 A fragments for this wave's rows stay register-resident.
  v16bf afrag[D_ / 32];
#pragma unroll
  for (int ks = 0; ks < D_ / 32; ++ks)
    afrag[ks] = load_frag_a(Qb, D_, t_tile, ks * 32);

  if (wid == 0)
    tdm_load_k_chunk(Kb, (unsigned)(size_t)&kstage[0][0], 0, 0);
  __syncthreads();

  for (int scI = 0; scI < L_ / SCH_; ++scI) {
    v8f acc[4] = {};                 // 4 independent accumulation chains
    for (int kcI = 0; kcI < D_ / KCH_; ++kcI) {
      int c = scI * (D_ / KCH_) + kcI;
      if (wid == 0) {
        if (c + 1 < NCH_) {
          int c2 = c + 1;
          tdm_load_k_chunk(Kb, (unsigned)(size_t)&kstage[c2 & 1][0],
                           c2 >> 2, c2 & 3);
          __builtin_amdgcn_s_wait_tensorcnt(1);   // oldest (chunk c) complete
        } else {
          __builtin_amdgcn_s_wait_tensorcnt(0);
        }
      }
      __syncthreads();               // chunk c visible to all waves
      const unsigned char* buf = &kstage[c & 1][0];

      // 16 wmma per chunk, software-pipelined 1 deep so the next B-fragment
      // ds_read_b128 pair is in flight while the current wmma issues.
      v16bf cur = load_frag_b_lds(buf, par, 0);
      v16bf nxt;
#pragma unroll
      for (int idx = 0; idx < (KCH_ / 32) * 4; ++idx) {
        int ks = idx >> 2, i = idx & 3;
        if (idx + 1 < (KCH_ / 32) * 4) {
          int idx2 = idx + 1;
          nxt = load_frag_b_lds(buf, (idx2 & 3) * 2 + par, idx2 >> 2);
        }
        acc[i] = wmma_bf16(afrag[kcI * (KCH_ / 32) + ks], cur, acc[i]);
        cur = nxt;
      }
      __syncthreads();               // all done with buf before it is reloaded
    }
    // scatter the 4 finished 16x16 tiles into the tau histogram
#pragma unroll
    for (int i = 0; i < 4; ++i) {
      int s = scI * SCH_ + (i * 2 + par) * 16 + n;
#pragma unroll
      for (int r = 0; r < 8; ++r) {
        int t = t_tile + rbase + r;
        int tau = (t - s) & (L_ - 1);
        atomicAdd(&sc[tau], acc[i][r]);          // ds_add_f32
      }
    }
  }
  __syncthreads();
  const float scale = 1.0f / (float)D_;
  for (int i = threadIdx.x; i < L_; i += 256)
    atomicAdd(&mean_corr[(size_t)b * L_ + i], sc[i] * scale);
}

// ---------------- top-k (k=8) + softmax, one WG per batch --------------------
__global__ __launch_bounds__(256) void topk_softmax_kernel(
    const float* __restrict__ mean_corr, float* __restrict__ wts,
    int* __restrict__ dly) {
  __shared__ float vals[L_];
  __shared__ float rv[256];
  __shared__ int   ri[256];
  __shared__ float topv[TOPK_];
  __shared__ int   topi[TOPK_];
  int b = blockIdx.x;
  int tid = threadIdx.x;
  for (int i = tid; i < L_; i += 256) vals[i] = mean_corr[(size_t)b * L_ + i];
  __syncthreads();

  for (int it = 0; it < TOPK_; ++it) {
    float best = -3.4e38f; int bidx = 0;
    for (int i = tid; i < L_; i += 256) {
      float v = vals[i];
      if (v > best) { best = v; bidx = i; }
    }
    rv[tid] = best; ri[tid] = bidx;
    __syncthreads();
    for (int s = 128; s > 0; s >>= 1) {
      if (tid < s) {
        float v2 = rv[tid + s]; int i2 = ri[tid + s];
        if (v2 > rv[tid] || (v2 == rv[tid] && i2 < ri[tid])) {
          rv[tid] = v2; ri[tid] = i2;
        }
      }
      __syncthreads();
    }
    if (tid == 0) { topv[it] = rv[0]; topi[it] = ri[0]; vals[ri[0]] = -3.4e38f; }
    __syncthreads();
  }

  if (tid == 0) {
    float mx = topv[0];
    for (int i = 1; i < TOPK_; ++i) mx = fmaxf(mx, topv[i]);
    float e[TOPK_]; float den = 0.0f;
    for (int i = 0; i < TOPK_; ++i) { e[i] = expf(topv[i] - mx); den += e[i]; }
    for (int i = 0; i < TOPK_; ++i) {
      wts[b * TOPK_ + i] = e[i] / den;
      dly[b * TOPK_ + i] = topi[i];
    }
  }
}

// ---------------- delay-gather + weighted aggregation ------------------------
__global__ __launch_bounds__(256) void agg_kernel(
    const unsigned short* __restrict__ V, const float* __restrict__ wts,
    const int* __restrict__ dly, unsigned short* __restrict__ ctx) {
  const int gpr = D_ / 8;
  int gid = blockIdx.x * 256 + threadIdx.x;
  int g = gid % gpr;
  int l = (gid / gpr) % L_;
  int b = gid / (gpr * L_);

  float w[TOPK_]; int de[TOPK_];
#pragma unroll
  for (int i = 0; i < TOPK_; ++i) {
    w[i] = wts[b * TOPK_ + i];
    de[i] = dly[b * TOPK_ + i];
  }
  float acc[8];
#pragma unroll
  for (int j = 0; j < 8; ++j) acc[j] = 0.0f;
#pragma unroll
  for (int i = 0; i < TOPK_; ++i) {
    int src = (l + de[i]) & (L_ - 1);
    uint4 raw = *(const uint4*)(V + (size_t)(b * L_ + src) * D_ + g * 8);
    const unsigned short* hs = (const unsigned short*)&raw;
#pragma unroll
    for (int j = 0; j < 8; ++j) acc[j] += w[i] * bf2f(hs[j]);
  }
  uint4 out;
  unsigned short* os = (unsigned short*)&out;
#pragma unroll
  for (int j = 0; j < 8; ++j) os[j] = f2bf(acc[j]);
  *(uint4*)(ctx + (size_t)gid * 8) = out;
}

// ---------------- launcher ----------------------------------------------------
extern "C" void kernel_launch(void* const* d_in, const int* in_sizes, int n_in,
                              void* d_out, int out_size, void* d_ws, size_t ws_size,
                              hipStream_t stream) {
  (void)in_sizes; (void)n_in; (void)out_size; (void)ws_size;
  const float* hidden = (const float*)d_in[0];
  const float* Wq = (const float*)d_in[1]; const float* bq = (const float*)d_in[2];
  const float* Wk = (const float*)d_in[3]; const float* bk = (const float*)d_in[4];
  const float* Wv = (const float*)d_in[5]; const float* bv = (const float*)d_in[6];
  const float* Wo = (const float*)d_in[7]; const float* bo = (const float*)d_in[8];

  char* ws = (char*)d_ws;
  size_t off = 0;
  const size_t XB = (size_t)BL_ * D_ * sizeof(unsigned short);   // 32 MiB
  const size_t WB = (size_t)D_ * D_ * sizeof(unsigned short);    // 512 KiB
  unsigned short* xbf = (unsigned short*)(ws + off); off += XB;  // X, reused as ctx
  unsigned short* qbf = (unsigned short*)(ws + off); off += XB;
  unsigned short* kbf = (unsigned short*)(ws + off); off += XB;
  unsigned short* vbf = (unsigned short*)(ws + off); off += XB;
  unsigned short* wqb = (unsigned short*)(ws + off); off += WB;
  unsigned short* wkb = (unsigned short*)(ws + off); off += WB;
  unsigned short* wvb = (unsigned short*)(ws + off); off += WB;
  unsigned short* wob = (unsigned short*)(ws + off); off += WB;
  float* corr = (float*)(ws + off); off += (size_t)B_ * L_ * sizeof(float);
  float* wts  = (float*)(ws + off); off += (size_t)B_ * TOPK_ * sizeof(float);
  int*   dly  = (int*)(ws + off);   off += (size_t)B_ * TOPK_ * sizeof(int);

  // Stage 0: precision conversion + corr zeroing
  cvt_f32_bf16_kernel<<<8192, 256, 0, stream>>>(hidden, xbf, BL_ * D_);
  cvt_f32_bf16_kernel<<<512, 256, 0, stream>>>(Wq, wqb, D_ * D_);
  cvt_f32_bf16_kernel<<<512, 256, 0, stream>>>(Wk, wkb, D_ * D_);
  cvt_f32_bf16_kernel<<<512, 256, 0, stream>>>(Wv, wvb, D_ * D_);
  cvt_f32_bf16_kernel<<<512, 256, 0, stream>>>(Wo, wob, D_ * D_);
  zero_f32_kernel<<<128, 256, 0, stream>>>(corr, B_ * L_);

  // Stage 1: Q/K/V projections (WMMA)
  dim3 ggrid(BL_ / 128, D_ / 64);
  gemm_bf16_kernel<0><<<ggrid, 256, 0, stream>>>(xbf, wqb, bq, qbf, BL_, D_, D_);
  gemm_bf16_kernel<0><<<ggrid, 256, 0, stream>>>(xbf, wkb, bk, kbf, BL_, D_, D_);
  gemm_bf16_kernel<0><<<ggrid, 256, 0, stream>>>(xbf, wvb, bv, vbf, BL_, D_, D_);

  // Stage 2: autocorrelation as Q.K^T Gram + anti-diagonal reduction
  //          (WMMA + TDM double-buffered K staging)
  corr_kernel<<<dim3(L_ / 64, B_), 256, 0, stream>>>(qbf, kbf, corr);

  // Stage 3: per-batch top-8 delays + softmax
  topk_softmax_kernel<<<B_, 256, 0, stream>>>(corr, wts, dly);

  // Stage 4: circular delay-gather + weighted sum -> ctx (reuses xbf)
  agg_kernel<<<(BL_ * D_ / 8) / 256, 256, 0, stream>>>(vbf, wts, dly, xbf);

  // Stage 5: output projection -> fp32 d_out (WMMA)
  gemm_bf16_kernel<1><<<ggrid, 256, 0, stream>>>(xbf, wob, bo, (float*)d_out,
                                                 BL_, D_, D_);
}
